// RGAT_53412213293589
// MI455X (gfx1250) — matrix operand order repaired and checked by reading, not compiled, not run on hardware
//
#include <hip/hip_runtime.h>
#include <math.h>

typedef __attribute__((ext_vector_type(16))) __bf16 v16bf;
typedef __attribute__((ext_vector_type(8)))  __bf16 v8bf;
typedef __attribute__((ext_vector_type(8)))  float  v8f;
typedef __attribute__((ext_vector_type(4)))  unsigned int v4u;
typedef __attribute__((ext_vector_type(8)))  int v8i;
typedef __attribute__((ext_vector_type(4)))  int v4i;

#define NEG_SLOPE 0.2f

#if defined(__gfx1250__) && __has_builtin(__builtin_amdgcn_tensor_load_to_lds) && \
    __has_builtin(__builtin_amdgcn_s_wait_tensorcnt)
#define USE_TDM 1
#else
#define USE_TDM 0
#endif

// ---------------------------------------------------------------- fill
__global__ void fill_kernel(float* __restrict__ p, float v, int n) {
  int i = blockIdx.x * blockDim.x + threadIdx.x;
  if (i < n) p[i] = v;
}

// ---------------------------------------------------------------- per-relation GEMM
// xw[n, r, :] = x[n, :] @ W[r]   (bf16 WMMA, f32 accumulate)
// grid = (N/16, R), block = 32*(HO/16); one wave per 16x16 C tile, K-step 64 (2 WMMA/iter).
// LDS: As   bf16[16][80]         (A tile, row-major, padded -> 16B-aligned b128 frag loads)
//      Bst  bf16[HO][80]         (B tile, transposed,  -> 16B-aligned b128 frag loads)
//      Wst  f32 [64][HO]         (TDM staging for raw weight tile)
__global__ void xw_gemm_kernel(const float* __restrict__ X,
                               const float* __restrict__ W,
                               float* __restrict__ XW,
                               int N, int R, int CIN, int HO) {
  const int r    = blockIdx.y;
  const int wave = threadIdx.x >> 5;
  const int lane = threadIdx.x & 31;
  const int col0 = wave << 4;
  const int row0 = blockIdx.x << 4;
  const int ln15 = lane & 15;
  const int half = lane >> 4;

  extern __shared__ char smem[];
  __bf16* As  = (__bf16*)smem;                      // 16*80*2 = 2560 B
  __bf16* Bst = (__bf16*)(smem + 2560);             // HO*80*2 B
  float*  Wst = (float*)(smem + 2560 + (size_t)HO * 160);  // 64*HO*4 B

  const float* Wr = W + (size_t)r * CIN * HO;

  v8f acc = {};
  for (int k0 = 0; k0 < CIN; k0 += 64) {
#if USE_TDM
    // ---- TDM: DMA raw f32 weight tile [64][HO] into LDS staging (wave 0 issues) ----
    if (wave == 0) {
      unsigned long long ga = (unsigned long long)(const void*)(Wr + (size_t)k0 * HO);
      unsigned ldsoff = (unsigned)(uintptr_t)(void*)Wst;  // low 32 bits of shared addr = LDS offset
      v4u g0 = { 1u,                                   // count=1, user mode
                 ldsoff,                               // lds_addr
                 (unsigned)ga,                         // global_addr[31:0]
                 (unsigned)(ga >> 32) | 0x80000000u }; // global_addr[56:32] | type=2
      v8i g1;
      g1[0] = 0x00020000;                        // data_size = 2 (4 bytes)
      g1[1] = (int)((unsigned)HO << 16);         // tensor_dim0 = HO
      g1[2] = (int)(((unsigned)(CIN - k0) & 0xFFFFu) << 16); // tensor_dim1 = rows left
      g1[3] = (int)((unsigned)HO << 16);         // tile_dim0 = HO
      g1[4] = 64;                                // tile_dim1 = 64 rows
      g1[5] = HO;                                // tensor_dim0_stride = HO
      g1[6] = 0;
      g1[7] = 0;
      v4i z = { 0, 0, 0, 0 };
#if __clang_major__ >= 23
      v8i z8 = { 0, 0, 0, 0, 0, 0, 0, 0 };
      __builtin_amdgcn_tensor_load_to_lds(g0, g1, z, z, z8, 0);
#else
      __builtin_amdgcn_tensor_load_to_lds(g0, g1, z, z, 0);
#endif
    }
#endif
    // ---- stage A tile 16x64 f32 -> bf16 LDS (float4 global loads) ----
    for (int c = threadIdx.x; c < 16 * 16; c += blockDim.x) {
      int m = c >> 4, k4 = (c & 15) << 2;
      const float4 v = *(const float4*)(X + (size_t)(row0 + m) * CIN + k0 + k4);
      __bf16* d = As + m * 80 + k4;
      d[0] = (__bf16)v.x; d[1] = (__bf16)v.y; d[2] = (__bf16)v.z; d[3] = (__bf16)v.w;
    }
#if USE_TDM
    if (wave == 0) __builtin_amdgcn_s_wait_tensorcnt(0);
    __syncthreads();          // Wst (TDM) + As visible
    // ---- LDS transpose + convert: Wst[kk][n] -> Bst[n][kk] (bf16) ----
    for (int i = threadIdx.x; i < 64 * HO; i += blockDim.x) {
      int kk = i / HO, n = i - kk * HO;
      Bst[n * 80 + kk] = (__bf16)Wst[i];
    }
#else
    // ---- fallback: stage B transposed directly from global ----
    for (int i = threadIdx.x; i < 64 * HO; i += blockDim.x) {
      int kk = i / HO, n = i - kk * HO;
      Bst[n * 80 + kk] = (__bf16)Wr[(size_t)(k0 + kk) * HO + n];
    }
    if (k0 + 64 < CIN)  // gfx1250 global_prefetch_b8 on next weight tile
      __builtin_prefetch(&Wr[(size_t)(k0 + 64) * HO + threadIdx.x], 0, 0);
#endif
    __syncthreads();          // As + Bst ready

    // ---- two 16x16x32 WMMAs over this K=64 slab ----
    #pragma unroll
    for (int w = 0; w < 2; ++w) {
      // A frag (16-bit A 16x32): lane(m=ln15,half): K = w*32 + half*8+e  /  +16
      const __bf16* Ab = As + ln15 * 80 + w * 32 + half * 8;
      v8bf a0 = *(const v8bf*)(Ab);
      v8bf a1 = *(const v8bf*)(Ab + 16);
      // B frag (16-bit B 32x16): lane(n=ln15,half): K = w*32 + half*8+e  /  +16
      const __bf16* Bb = Bst + (size_t)(col0 + ln15) * 80 + w * 32 + half * 8;
      v8bf b0 = *(const v8bf*)(Bb);
      v8bf b1 = *(const v8bf*)(Bb + 16);
      v16bf a = __builtin_shufflevector(a0, a1, 0,1,2,3,4,5,6,7,8,9,10,11,12,13,14,15);
      v16bf b = __builtin_shufflevector(b0, b1, 0,1,2,3,4,5,6,7,8,9,10,11,12,13,14,15);
      acc = __builtin_amdgcn_wmma_f32_16x16x32_bf16(false, a, false, b,
                                                    (short)0, acc, false, false);
    }
    __syncthreads();          // before LDS reuse
  }

  // C layout: VGPR v -> M = v + 8*half, N = ln15
  #pragma unroll
  for (int v = 0; v < 8; ++v) {
    int m = v + half * 8;
    XW[((size_t)(row0 + m) * R + r) * HO + (col0 + ln15)] = acc[v];
  }
}

// ---------------------------------------------------------------- lee = le @ e  [ED x H]
__global__ void lee_kernel(const float* __restrict__ le, const float* __restrict__ ev,
                           float* __restrict__ lee, int ED, int HO, int H) {
  int i = threadIdx.x;
  if (i >= ED * H) return;
  int d = i / H, h = i % H;
  float s = 0.f;
  for (int c = 0; c < HO; ++c) s += le[(size_t)d * HO + c] * ev[(size_t)c * H + h];
  lee[i] = s;
}

// ---------------------------------------------------------------- aq/ak per (node, relation, head)
__global__ void attn_node_kernel(const float* __restrict__ XW,
                                 const float* __restrict__ q, const float* __restrict__ k,
                                 float* __restrict__ AQ, float* __restrict__ AK,
                                 int NRH, int HO, int H) {
  int idx = blockIdx.x * blockDim.x + threadIdx.x;
  if (idx >= NRH) return;
  int h = idx % H, nr = idx / H;
  const float* row = XW + (size_t)nr * HO;
  float sq = 0.f, sk = 0.f;
  for (int c = 0; c < HO; ++c) {
    float xv = row[c];
    sq += xv * q[(size_t)c * H + h];
    sk += xv * k[(size_t)c * H + h];
  }
  AQ[idx] = sq;
  AK[idx] = sk;
}

// ---------------------------------------------------------------- alpha + segment max
__global__ void edge_alpha_kernel(const int* __restrict__ src, const int* __restrict__ dst,
                                  const int* __restrict__ et, const float* __restrict__ eattr,
                                  const float* __restrict__ lee,
                                  const float* __restrict__ AQ, const float* __restrict__ AK,
                                  float* __restrict__ alpha, float* __restrict__ amax,
                                  int E, int N, int R, int H, int ED) {
  int idx = blockIdx.x * blockDim.x + threadIdx.x;
  if (idx >= E * H) return;
  int h = idx % H, e = idx / H;
  int s = src[e], d = dst[e], t = et[e];
  float a = AQ[((size_t)d * R + t) * H + h] + AK[((size_t)s * R + t) * H + h];
  for (int c = 0; c < ED; ++c) a += eattr[(size_t)e * ED + c] * lee[c * H + h];
  a = (a > 0.f) ? a : NEG_SLOPE * a;          // leaky_relu
  alpha[idx] = a;
  unsafeAtomicMax(&amax[((size_t)t * N + d) * H + h], a);   // global_atomic_max_num_f32
}

// ---------------------------------------------------------------- exp + segment sum
__global__ void edge_exp_kernel(const int* __restrict__ dst, const int* __restrict__ et,
                                float* __restrict__ alpha, const float* __restrict__ amax,
                                float* __restrict__ den, int E, int N, int H) {
  int idx = blockIdx.x * blockDim.x + threadIdx.x;
  if (idx >= E * H) return;
  int h = idx % H, e = idx / H;
  int d = dst[e], t = et[e];
  size_t seg = ((size_t)t * N + d) * H + h;
  float ex = __expf(alpha[idx] - amax[seg]);
  alpha[idx] = ex;
  unsafeAtomicAdd(&den[seg], ex);             // global_atomic_add_f32
}

// ---------------------------------------------------------------- weighted scatter-add
__global__ void edge_agg_kernel(const int* __restrict__ src, const int* __restrict__ dst,
                                const int* __restrict__ et,
                                const float* __restrict__ alpha, const float* __restrict__ den,
                                const float* __restrict__ XW, float* __restrict__ agg,
                                int E, int N, int R, int HO, int H) {
  long long idx = (long long)blockIdx.x * blockDim.x + threadIdx.x;
  long long tot = (long long)E * HO;
  if (idx >= tot) return;
  int c = (int)(idx % HO);
  int e = (int)(idx / HO);
  int s = src[e], d = dst[e], t = et[e];
  int O = HO / H;
  int h = c / O;
  float ex = alpha[(size_t)e * H + h];
  float dn = den[((size_t)t * N + d) * H + h];
  float att = ex / (dn + 1e-16f);
  float xj = XW[((size_t)s * R + t) * HO + c];
  unsafeAtomicAdd(&agg[(size_t)d * HO + c], att * xj);
}

// ---------------------------------------------------------------- head mean + bias (+relu)
__global__ void finalize_kernel(const float* __restrict__ agg, const float* __restrict__ b,
                                float* __restrict__ out, int N, int H, int O, int do_relu) {
  int idx = blockIdx.x * blockDim.x + threadIdx.x;
  if (idx >= N * O) return;
  int c = idx % O, n = idx / O;
  float s = 0.f;
  for (int h = 0; h < H; ++h) s += agg[(size_t)n * H * O + h * O + c];
  float v = s / (float)H + b[c];
  if (do_relu) v = fmaxf(v, 0.f);
  out[idx] = v;
}

// ================================================================ host driver
static void run_layer(const float* x, int cin, int cout, int heads,
                      int N, int E, int R,
                      const int* src, const int* dst, const int* et,
                      const float* eattr,
                      const float* W, const float* q, const float* k,
                      const float* le, const float* ev, const float* b,
                      float* xw, float* aq, float* ak, float* lee,
                      float* alpha, float* amax, float* den, float* agg,
                      float* out, int do_relu, hipStream_t stream) {
  const int HO = heads * cout;
  const int H  = heads;
  const int ED = 16;

  // init accumulators
  {
    int n = N * HO;
    fill_kernel<<<(n + 255) / 256, 256, 0, stream>>>(agg, 0.f, n);
    n = R * N * H;
    fill_kernel<<<(n + 255) / 256, 256, 0, stream>>>(amax, -1e30f, n);
    fill_kernel<<<(n + 255) / 256, 256, 0, stream>>>(den, 0.f, n);
  }
  // projection GEMM (WMMA, K-step 64, TDM-staged weights)
  {
    dim3 grid(N / 16, R);
    int block = 32 * (HO / 16);
    size_t shm = 2560 + (size_t)HO * 160 + (size_t)64 * HO * 4;
    xw_gemm_kernel<<<grid, block, shm, stream>>>(x, W, xw, N, R, cin, HO);
  }
  // lee = le @ e
  lee_kernel<<<1, 64, 0, stream>>>(le, ev, lee, ED, HO, H);
  // aq / ak
  {
    int n = N * R * H;
    attn_node_kernel<<<(n + 255) / 256, 256, 0, stream>>>(xw, q, k, aq, ak, n, HO, H);
  }
  // alpha + segment max, then exp + segment sum
  {
    int n = E * H;
    edge_alpha_kernel<<<(n + 255) / 256, 256, 0, stream>>>(src, dst, et, eattr, lee,
                                                           aq, ak, alpha, amax,
                                                           E, N, R, H, ED);
    edge_exp_kernel<<<(n + 255) / 256, 256, 0, stream>>>(dst, et, alpha, amax, den, E, N, H);
  }
  // aggregate
  {
    long long tot = (long long)E * HO;
    unsigned int blocks = (unsigned int)((tot + 255) / 256);
    edge_agg_kernel<<<blocks, 256, 0, stream>>>(src, dst, et, alpha, den, xw, agg,
                                                E, N, R, HO, H);
  }
  // mean heads + bias (+relu)
  {
    int n = N * cout;
    finalize_kernel<<<(n + 255) / 256, 256, 0, stream>>>(agg, b, out, N, H, cout, do_relu);
  }
}

extern "C" void kernel_launch(void* const* d_in, const int* in_sizes, int n_in,
                              void* d_out, int out_size, void* d_ws, size_t ws_size,
                              hipStream_t stream) {
  (void)n_in; (void)out_size; (void)ws_size;
  const int IN = 128, HID = 64, OUT = 64, R = 5;
  const int N = in_sizes[0] / IN;
  const int E = in_sizes[1] / 2;

  const float* x     = (const float*)d_in[0];
  const int*   ei    = (const int*)d_in[1];
  const int*   src   = ei;
  const int*   dst   = ei + E;
  const int*   et    = (const int*)d_in[2];
  const float* eattr = (const float*)d_in[3];

  const float* W[3]  = { (const float*)d_in[4],  (const float*)d_in[10], (const float*)d_in[16] };
  const float* q[3]  = { (const float*)d_in[5],  (const float*)d_in[11], (const float*)d_in[17] };
  const float* k[3]  = { (const float*)d_in[6],  (const float*)d_in[12], (const float*)d_in[18] };
  const float* le[3] = { (const float*)d_in[7],  (const float*)d_in[13], (const float*)d_in[19] };
  const float* ev[3] = { (const float*)d_in[8],  (const float*)d_in[14], (const float*)d_in[20] };
  const float* bi[3] = { (const float*)d_in[9],  (const float*)d_in[15], (const float*)d_in[21] };

  // workspace layout (floats)
  float* ws = (float*)d_ws;
  size_t off = 0;
  float* xw    = ws + off; off += (size_t)N * R * 128;   // max HO = 128
  float* aq    = ws + off; off += (size_t)N * R * 2;
  float* ak    = ws + off; off += (size_t)N * R * 2;
  float* lee   = ws + off; off += 64;
  float* alpha = ws + off; off += (size_t)E * 2;
  float* amax  = ws + off; off += (size_t)R * N * 2;
  float* den   = ws + off; off += (size_t)R * N * 2;
  float* agg   = ws + off; off += (size_t)N * 128;
  float* h1    = ws + off; off += (size_t)N * HID;
  float* h2    = ws + off; off += (size_t)N * HID;

  // layer 1: 128 -> 64, heads 2, relu
  run_layer(x, IN, HID, 2, N, E, R, src, dst, et, eattr,
            W[0], q[0], k[0], le[0], ev[0], bi[0],
            xw, aq, ak, lee, alpha, amax, den, agg, h1, 1, stream);
  // layer 2: 64 -> 64, heads 2, relu
  run_layer(h1, HID, HID, 2, N, E, R, src, dst, et, eattr,
            W[1], q[1], k[1], le[1], ev[1], bi[1],
            xw, aq, ak, lee, alpha, amax, den, agg, h2, 1, stream);
  // layer 3: 64 -> 64, heads 1, no relu, output
  run_layer(h2, HID, OUT, 1, N, E, R, src, dst, et, eattr,
            W[2], q[2], k[2], le[2], ev[2], bi[2],
            xw, aq, ak, lee, alpha, amax, den, agg, (float*)d_out, 0, stream);
}
